// WindowAttention_Kai_6107443495138
// MI455X (gfx1250) — compile-verified, matrix operand order checked by reading, not run on hardware
//
#include <hip/hip_runtime.h>
#include <hip/hip_bf16.h>

typedef __bf16 bf16_t;
typedef __attribute__((ext_vector_type(16))) __bf16 v16bf;
typedef __attribute__((ext_vector_type(8)))  __bf16 v8bf;
typedef __attribute__((ext_vector_type(8)))  float  v8f;

#define WINSZ   7
#define NTOK    49
#define NPAD    64
#define CH      192
#define NHEAD   6
#define DHEAD   32
#define THREADS 256
#define NWAVE   8
#define HFEAT   112
#define NFEAT   (HFEAT * HFEAT)

// ---------------------------------------------------------------------------
// LDS element counts
#define X_ELEMS    (NPAD * CH)          // 12288 bf16  (aliased as attnOut later)
#define QK_ELEMS   (NHEAD * NPAD * DHEAD) // 12288 bf16 each for q, k, vT
#define SC_ELEMS   (NPAD * NPAD)        // 4096 f32 scores
#define AT_ELEMS   (NPAD * NPAD)        // 4096 bf16 attn probs
#define NBIAS      (3 * CH + CH + (2 * WINSZ - 1) * (2 * WINSZ - 1) * NHEAD) // 576+192+1014
#define SMEM_BYTES (2 * (X_ELEMS + 3 * QK_ELEMS + AT_ELEMS) + 4 * (SC_ELEMS + NBIAS) + 64)

// ---------------------------------------------------------------------------
// Load a 16x32 bf16 WMMA operand fragment from a row-major matrix.
// A-matrix: "row" = M index.  B-matrix: "row" = N index (column of B = row of
// the row-major source, K contiguous) — same striping, per ISA 7.12.2:
//   lanes 0-15 : row = r0+lane,     K = c0+[0..7] and c0+[16..23]
//   lanes16-31 : row = r0+lane-16,  K = c0+[8..15] and c0+[24..31]
__device__ __forceinline__ v16bf load_frag(const bf16_t* __restrict__ base,
                                           int ld, int r0, int c0, int lane) {
    int r = r0 + (lane & 15);
    int c = c0 + ((lane >> 4) << 3);
    const v8bf* p0 = (const v8bf*)(base + r * ld + c);
    const v8bf* p1 = (const v8bf*)(base + r * ld + c + 16);
    v8bf lo = *p0, hi = *p1;
    v16bf f;
#pragma unroll
    for (int i = 0; i < 8; ++i) { f[i] = lo[i]; f[i + 8] = hi[i]; }
    return f;
}

// ---------------------------------------------------------------------------
// Prologue: fp32 weights -> bf16 in workspace (reused by all 2048 blocks via L2)
__global__ void wa_convert_weights(const float* __restrict__ qkv_w,
                                   const float* __restrict__ proj_w,
                                   bf16_t* __restrict__ ws) {
    const int NQ = 3 * CH * CH, NP = CH * CH;
    int i = blockIdx.x * blockDim.x + threadIdx.x;
    if (i < NQ)           ws[i] = (bf16_t)qkv_w[i];
    else if (i < NQ + NP) ws[i] = (bf16_t)proj_w[i - NQ];
}

// ---------------------------------------------------------------------------
// Fused per-window attention: one workgroup (8 wave32) per 7x7 window.
__global__ __launch_bounds__(THREADS)
void wa_fused_window_attn(const float* __restrict__ x,
                          const float* __restrict__ qkv_b,
                          const float* __restrict__ proj_b,
                          const float* __restrict__ rpb,
                          const bf16_t* __restrict__ qkvW,   // [576][192] bf16
                          const bf16_t* __restrict__ projW,  // [192][192] bf16
                          float* __restrict__ out) {
    extern __shared__ char smem_raw[];
    bf16_t* xBuf    = (bf16_t*)smem_raw;            // [64][192] window tokens
    bf16_t* qLds    = xBuf   + X_ELEMS;             // [6][64][32]
    bf16_t* kLds    = qLds   + QK_ELEMS;            // [6][64][32]
    bf16_t* vTLds   = kLds   + QK_ELEMS;            // [6][32][64] (transposed)
    bf16_t* attnBuf = vTLds  + QK_ELEMS;            // [64][64] probs (bf16)
    float*  scores  = (float*)(attnBuf + AT_ELEMS); // [64][64] f32
    float*  biasQ   = scores + SC_ELEMS;            // [576]
    float*  biasP   = biasQ + 3 * CH;               // [192]
    float*  rpbL    = biasP + CH;                   // [169*6]
    bf16_t* attnOut = xBuf;                         // alias: [64][192] after QKV

    const int tid  = threadIdx.x;
    const int lane = tid & 31;
    const int wave = tid >> 5;
    const int win  = blockIdx.x;        // 0..2047
    const int b    = win >> 8;
    const int wy   = (win >> 4) & 15;
    const int wx   = win & 15;
    const float scale = 0.17677669529663687f;   // 32^-0.5

    // ---- stage biases + relative-position table into LDS -----------------
    for (int i = tid; i < NBIAS; i += THREADS) {
        float v;
        if (i < 3 * CH)          v = qkv_b[i];
        else if (i < 4 * CH)     v = proj_b[i - 3 * CH];
        else                     v = rpb[i - 4 * CH];
        biasQ[i] = v;
    }

    // ---- gather window tokens (fp32 -> bf16), zero-pad rows 49..63 -------
    for (int e = tid; e < X_ELEMS; e += THREADS) {
        int row = e / CH;
        int ch  = e - row * CH;
        float v = 0.f;
        if (row < NTOK) {
            int tok = (wy * WINSZ + row / WINSZ) * HFEAT + (wx * WINSZ + row % WINSZ);
            v = x[((size_t)b * NFEAT + tok) * CH + ch];
        }
        xBuf[e] = (bf16_t)v;
    }
    __syncthreads();

    // ---- QKV GEMM: [64,192] x [192,576]  (4 x 36 tiles, 6 K-steps) -------
    for (int t = wave; t < 4 * 36; t += NWAVE) {
        int mi = t / 36, ni = t % 36;
        v8f acc = {};
#pragma unroll
        for (int ks = 0; ks < 6; ++ks) {
            v16bf a  = load_frag(xBuf, CH, mi * 16, ks * 32, lane);
            v16bf bm = load_frag(qkvW, CH, ni * 16, ks * 32, lane);
            acc = __builtin_amdgcn_wmma_f32_16x16x32_bf16(
                      false, a, false, bm, (short)0, acc, false, false);
        }
        // epilogue: +bias, route into q/k (row-major) or v (transposed)
        int ob   = ni * 16;
        float bv = biasQ[ob + (lane & 15)];
        int which = ob / CH;
        int rem   = ob - which * CH;
        int head  = rem / DHEAD;
        int part  = rem - head * DHEAD;          // 0 or 16
        int cc    = part + (lane & 15);
        int rr    = mi * 16 + ((lane >> 4) << 3);
        if (which == 2) {
            bf16_t* dst = vTLds + head * (DHEAD * NPAD);
#pragma unroll
            for (int i = 0; i < 8; ++i)
                dst[cc * NPAD + rr + i] = (bf16_t)(acc[i] + bv);
        } else {
            bf16_t* dst = (which == 0 ? qLds : kLds) + head * (NPAD * DHEAD);
#pragma unroll
            for (int i = 0; i < 8; ++i)
                dst[(rr + i) * DHEAD + cc] = (bf16_t)(acc[i] + bv);
        }
    }
    __syncthreads();

    // ---- per-head attention ---------------------------------------------
    for (int h = 0; h < NHEAD; ++h) {
        const bf16_t* qh = qLds  + h * NPAD * DHEAD;
        const bf16_t* kh = kLds  + h * NPAD * DHEAD;
        const bf16_t* vh = vTLds + h * DHEAD * NPAD;

        // S = (q kᵀ) * scale : d=32 -> single WMMA per 16x16 tile
        for (int t = wave; t < 16; t += NWAVE) {
            int mi = t >> 2, nj = t & 3;
            v16bf a  = load_frag(qh, DHEAD, mi * 16, 0, lane);
            v16bf bm = load_frag(kh, DHEAD, nj * 16, 0, lane);
            v8f acc = {};
            acc = __builtin_amdgcn_wmma_f32_16x16x32_bf16(
                      false, a, false, bm, (short)0, acc, false, false);
            int cc = nj * 16 + (lane & 15);
            int rr = mi * 16 + ((lane >> 4) << 3);
#pragma unroll
            for (int i = 0; i < 8; ++i)
                scores[(rr + i) * NPAD + cc] = acc[i] * scale;
        }
        __syncthreads();

        // softmax over j<49 with relative-position bias; zero pad columns
        if (tid < NPAD) {
            int i = tid;
            if (i < NTOK) {
                int yi = i / WINSZ, xi = i % WINSZ;
                float m = -3.0e38f;
                for (int j = 0; j < NTOK; ++j) {
                    int yj = j / WINSZ, xj = j % WINSZ;
                    int idx = (yi - yj + WINSZ - 1) * (2 * WINSZ - 1)
                            + (xi - xj + WINSZ - 1);
                    float s = scores[i * NPAD + j] + rpbL[idx * NHEAD + h];
                    scores[i * NPAD + j] = s;
                    m = fmaxf(m, s);
                }
                float sum = 0.f;
                for (int j = 0; j < NTOK; ++j) {
                    float e = __expf(scores[i * NPAD + j] - m);
                    scores[i * NPAD + j] = e;
                    sum += e;
                }
                float inv = 1.0f / sum;
                for (int j = 0; j < NTOK; ++j)
                    attnBuf[i * NPAD + j] = (bf16_t)(scores[i * NPAD + j] * inv);
                for (int j = NTOK; j < NPAD; ++j)
                    attnBuf[i * NPAD + j] = (bf16_t)0.f;
            } else {
                for (int j = 0; j < NPAD; ++j)
                    attnBuf[i * NPAD + j] = (bf16_t)0.f;
            }
        }
        __syncthreads();

        // O_h = A @ V : [64,64] x [64,32] (4 x 2 tiles, 2 K-steps)
        for (int t = wave; t < 8; t += NWAVE) {
            int mi = t >> 1, nd = t & 1;
            v8f acc = {};
#pragma unroll
            for (int kb = 0; kb < 2; ++kb) {
                v16bf a  = load_frag(attnBuf, NPAD, mi * 16, kb * 32, lane);
                v16bf bm = load_frag(vh,      NPAD, nd * 16, kb * 32, lane);
                acc = __builtin_amdgcn_wmma_f32_16x16x32_bf16(
                          false, a, false, bm, (short)0, acc, false, false);
            }
            int cc = h * DHEAD + nd * 16 + (lane & 15);
            int rr = mi * 16 + ((lane >> 4) << 3);
#pragma unroll
            for (int i = 0; i < 8; ++i)
                attnOut[(rr + i) * CH + cc] = (bf16_t)acc[i];
        }
        __syncthreads();
    }

    // ---- proj GEMM + window-reverse scatter: [64,192] x [192,192] --------
    for (int t = wave; t < 4 * 12; t += NWAVE) {
        int mi = t / 12, ni = t % 12;
        v8f acc = {};
#pragma unroll
        for (int ks = 0; ks < 6; ++ks) {
            v16bf a  = load_frag(attnOut, CH, mi * 16, ks * 32, lane);
            v16bf bm = load_frag(projW,   CH, ni * 16, ks * 32, lane);
            acc = __builtin_amdgcn_wmma_f32_16x16x32_bf16(
                      false, a, false, bm, (short)0, acc, false, false);
        }
        int cc = ni * 16 + (lane & 15);
        int rr = mi * 16 + ((lane >> 4) << 3);
        float bv = biasP[cc];
#pragma unroll
        for (int i = 0; i < 8; ++i) {
            int row = rr + i;
            if (row < NTOK) {
                int tok = (wy * WINSZ + row / WINSZ) * HFEAT
                        + (wx * WINSZ + row % WINSZ);
                out[((size_t)b * NFEAT + tok) * CH + cc] = acc[i] + bv;
            }
        }
    }
}

// ---------------------------------------------------------------------------
extern "C" void kernel_launch(void* const* d_in, const int* in_sizes, int n_in,
                              void* d_out, int out_size, void* d_ws, size_t ws_size,
                              hipStream_t stream) {
    const float* x      = (const float*)d_in[0];
    const float* qkv_w  = (const float*)d_in[1];
    const float* qkv_b  = (const float*)d_in[2];
    const float* proj_w = (const float*)d_in[3];
    const float* proj_b = (const float*)d_in[4];
    const float* rpb    = (const float*)d_in[5];
    // d_in[6], d_in[7] = H_feat, W_feat (fixed at 112)

    bf16_t* wsBf = (bf16_t*)d_ws;
    const int NQ = 3 * CH * CH, NP = CH * CH;

    wa_convert_weights<<<(NQ + NP + 255) / 256, 256, 0, stream>>>(qkv_w, proj_w, wsBf);

    const bf16_t* qkvW  = wsBf;
    const bf16_t* projW = wsBf + NQ;

    // 8 batches * 16 * 16 windows = 2048 workgroups, 8 wave32 each
    wa_fused_window_attn<<<2048, THREADS, SMEM_BYTES, stream>>>(
        x, qkv_b, proj_b, rpb, qkvW, projW, (float*)d_out);
}
